// Attention_65206193488052
// MI455X (gfx1250) — matrix operand order repaired and checked by reading, not compiled, not run on hardware
//
#include <hip/hip_runtime.h>
#include <stdint.h>

// ---------------- types ----------------
typedef __bf16 bf16_t;
typedef __attribute__((ext_vector_type(16))) __bf16       v16bf;
typedef __attribute__((ext_vector_type(8)))  float        v8f;
typedef __attribute__((ext_vector_type(4)))  unsigned int u32x4;
typedef __attribute__((ext_vector_type(8)))  float        f32x8;

union FragU { v16bf v; u32x4 q[2]; };

#define T_SEQ 4096
#define DMODEL 1024

// ---------------- fp32 -> bf16 copy ----------------
__global__ __launch_bounds__(256) void cvt_f32_bf16(const float* __restrict__ in,
                                                    bf16_t* __restrict__ out, int n) {
  int i = blockIdx.x * 256 + threadIdx.x;
  if (i < n) out[i] = (bf16_t)in[i];
}

// ------------- fp32 [R,C] -> bf16 [C,R] transpose -------------
__global__ __launch_bounds__(256) void transpose_cvt(const float* __restrict__ in,
                                                     bf16_t* __restrict__ out,
                                                     int R, int C) {
  __shared__ float tile[32][33];
  int bx = blockIdx.x * 32;           // col base in input
  int by = blockIdx.y * 32;           // row base in input
  int tx = threadIdx.x & 31;
  int ty = threadIdx.x >> 5;          // 0..7
  #pragma unroll
  for (int i = ty; i < 32; i += 8)
    tile[i][tx] = in[(long)(by + i) * C + bx + tx];
  __syncthreads();
  #pragma unroll
  for (int i = ty; i < 32; i += 8)
    out[(long)(bx + i) * R + by + tx] = (bf16_t)tile[tx][i];
}

// ------------- NT GEMM: C[M,N] = A[M,K] * Bs[N,K]^T via WMMA bf16 -------------
// AMODE: 0 = A is bf16, 1 = A is f32 (converted on load)
// OMODE: 0 = store bf16, 1 = store f32, 2 = store f32 + causal mask (-inf above diag)
template <int AMODE, int OMODE>
__global__ __launch_bounds__(256) void gemm_nt_wmma(const void* __restrict__ Av,
                                                    const bf16_t* __restrict__ B,
                                                    void* __restrict__ Outv,
                                                    int M, int N, int K) {
  const int lane = threadIdx.x & 31;
  const int wave = threadIdx.x >> 5;
  const int wm   = wave & 1;          // 2 waves along M
  const int wn   = wave >> 1;         // 4 waves along N
  const int bm   = blockIdx.y * 128;
  const int bn   = blockIdx.x * 128;
  const int m0   = bm + wm * 64;
  const int n0   = bn + wn * 32;
  const int half = lane >> 4;         // 0 | 1
  const int l15  = lane & 15;
  const int koff = half * 8;

  v8f acc[4][2] = {};

  const bool skip_all = (OMODE == 2) && (bn > bm + 127);  // tile fully above diagonal

  if (!skip_all) {
    const bf16_t* Ab = (const bf16_t*)Av;
    const float*  Af = (const float*)Av;

    for (int k = 0; k < K; k += 32) {
      FragU fa[4];
      #pragma unroll
      for (int mt = 0; mt < 4; ++mt) {
        long rb = (long)(m0 + mt * 16 + l15) * K + k;
        if (AMODE == 0) {
          fa[mt].q[0] = *(const u32x4*)(Ab + rb + koff);
          fa[mt].q[1] = *(const u32x4*)(Ab + rb + koff + 16);
        } else {
          f32x8 c0 = *(const f32x8*)(Af + rb + koff);
          f32x8 c1 = *(const f32x8*)(Af + rb + koff + 16);
          #pragma unroll
          for (int i = 0; i < 8; ++i) {
            fa[mt].v[i]     = (bf16_t)c0[i];
            fa[mt].v[8 + i] = (bf16_t)c1[i];
          }
        }
      }
      FragU fb[2];
      #pragma unroll
      for (int nt = 0; nt < 2; ++nt) {
        const bf16_t* pb = B + (long)(n0 + nt * 16 + l15) * K + k + half * 8;
        fb[nt].q[0] = *(const u32x4*)pb;
        fb[nt].q[1] = *(const u32x4*)(pb + 16);
        __builtin_prefetch(pb + 32, 0, 1);  // next K-step (stays inside workspace)
      }
      #pragma unroll
      for (int mt = 0; mt < 4; ++mt)
        #pragma unroll
        for (int nt = 0; nt < 2; ++nt)
          acc[mt][nt] = __builtin_amdgcn_wmma_f32_16x16x32_bf16(
              false, fa[mt].v, false, fb[nt].v, (short)0, acc[mt][nt], false, false);
    }
  }

  // Epilogue: C/D layout — VGPR r holds row m0+mt*16+half*8+r, col n0+nt*16+l15
  #pragma unroll
  for (int mt = 0; mt < 4; ++mt) {
    #pragma unroll
    for (int nt = 0; nt < 2; ++nt) {
      #pragma unroll
      for (int r = 0; r < 8; ++r) {
        int row = m0 + mt * 16 + half * 8 + r;
        int col = n0 + nt * 16 + l15;
        float v = skip_all ? -__builtin_inff() : acc[mt][nt][r];
        if (OMODE == 2 && col > row) v = -__builtin_inff();
        if (OMODE == 0)
          ((bf16_t*)Outv)[(long)row * N + col] = (bf16_t)v;
        else
          ((float*)Outv)[(long)row * N + col] = v;
      }
    }
  }
}

// ------------- in-place row softmax on f32 [T,T] -------------
// Row (16 KB) is staged ONCE into LDS with CDNA5 async global->LDS copies
// (ASYNCcnt), then all three passes (max, sum-exp, normalize) read LDS.
__global__ __launch_bounds__(256) void softmax_rows(float* __restrict__ S, int T) {
  __shared__ float row_s[T_SEQ];   // 16 KB
  __shared__ float red[256];
  const int tid = threadIdx.x;
  float* p = S + (long)blockIdx.x * T;

  // async copy: each thread moves 16 B per step, 4 KB per block-step
  for (int base = 0; base < T; base += 256 * 4) {
    const float* g = p + base + tid * 4;
    unsigned loff = (unsigned)(uintptr_t)(&row_s[base + tid * 4]);  // LDS byte offset
    asm volatile("global_load_async_to_lds_b128 %0, %1, off"
                 :: "v"(loff), "v"(g) : "memory");
  }
  asm volatile("s_wait_asynccnt 0x0" ::: "memory");
  __syncthreads();

  float m = -__builtin_inff();
  for (int j = tid; j < T; j += 256) m = fmaxf(m, row_s[j]);
  red[tid] = m;
  __syncthreads();
  for (int s = 128; s > 0; s >>= 1) {
    if (tid < s) red[tid] = fmaxf(red[tid], red[tid + s]);
    __syncthreads();
  }
  const float mx = red[0];
  __syncthreads();

  float sum = 0.f;
  for (int j = tid; j < T; j += 256) sum += __expf(row_s[j] - mx);
  red[tid] = sum;
  __syncthreads();
  for (int s = 128; s > 0; s >>= 1) {
    if (tid < s) red[tid] += red[tid + s];
    __syncthreads();
  }
  const float inv = 1.0f / red[0];

  for (int j = tid; j < T; j += 256) p[j] = __expf(row_s[j] - mx) * inv;
}

// ---------------- launcher ----------------
extern "C" void kernel_launch(void* const* d_in, const int* in_sizes, int n_in,
                              void* d_out, int out_size, void* d_ws, size_t ws_size,
                              hipStream_t stream) {
  const float* x   = (const float*)d_in[0];   // [T, D]
  const float* Wqk = (const float*)d_in[1];   // [D, D]
  const float* Wov = (const float*)d_in[2];   // [D, D]
  float* out = (float*)d_out;                 // [T, D]

  char* ws = (char*)d_ws;
  // workspace layout (bytes)
  bf16_t* xb   = (bf16_t*)(ws + (size_t)0);          //  8 MB  bf16 x      [T,D]
  bf16_t* xT   = (bf16_t*)(ws + (size_t)(8  << 20)); //  8 MB  bf16 x^T    [D,T]
  bf16_t* WqkT = (bf16_t*)(ws + (size_t)(16 << 20)); //  2 MB  bf16 Wqk^T  [D,D]
  bf16_t* WovT = (bf16_t*)(ws + (size_t)(18 << 20)); //  2 MB  bf16 Wov^T  [D,D]
  bf16_t* qb   = (bf16_t*)(ws + (size_t)(20 << 20)); //  8 MB  bf16 q      [T,D]
  bf16_t* O1   = (bf16_t*)(ws + (size_t)(28 << 20)); //  8 MB  bf16 attn@x [T,D]
  float*  S    = (float*)(ws + (size_t)(36 << 20));  // 64 MB  f32 scores/attn [T,T]

  const int T = T_SEQ, D = DMODEL;

  // 1) precision conversion + transposes
  cvt_f32_bf16<<<(T * D) / 256, 256, 0, stream>>>(x, xb, T * D);
  transpose_cvt<<<dim3(D / 32, T / 32), 256, 0, stream>>>(x,   xT,   T, D);
  transpose_cvt<<<dim3(D / 32, D / 32), 256, 0, stream>>>(Wqk, WqkT, D, D);
  transpose_cvt<<<dim3(D / 32, D / 32), 256, 0, stream>>>(Wov, WovT, D, D);

  // 2) q = x @ Wqk            (bf16 out)
  gemm_nt_wmma<0, 0><<<dim3(D / 128, T / 128), 256, 0, stream>>>(xb, WqkT, qb, T, D, D);

  // 3) S = q @ x^T + causal   (f32 out, masked)
  gemm_nt_wmma<0, 2><<<dim3(T / 128, T / 128), 256, 0, stream>>>(qb, xb, S, T, T, D);

  // 4) attn = softmax(S)      (in place, f32)
  softmax_rows<<<T, 256, 0, stream>>>(S, T);

  // 5) O1 = attn @ x          (A = f32 attn, bf16 out)
  gemm_nt_wmma<1, 0><<<dim3(D / 128, T / 128), 256, 0, stream>>>(S, xT, O1, T, D, T);

  // 6) out = O1 @ Wov         (f32 out to d_out)
  gemm_nt_wmma<0, 1><<<dim3(D / 128, T / 128), 256, 0, stream>>>(O1, WovT, out, T, D, D);
}
